// node_learning_69277822484771
// MI455X (gfx1250) — matrix operand order repaired and checked by reading, not compiled, not run on hardware
//
#include <hip/hip_runtime.h>
#include <stdint.h>

typedef __attribute__((ext_vector_type(16))) __bf16 v16bf;
typedef __attribute__((ext_vector_type(2)))  __bf16 v2bf;
typedef __attribute__((ext_vector_type(8)))  float  v8f;
typedef __attribute__((ext_vector_type(4)))  unsigned int v4u;
typedef __attribute__((ext_vector_type(8)))  int          v8i_;
typedef __attribute__((ext_vector_type(4)))  int          v4i_;

#define NN    10000            // nodes
#define CDIM  128              // in/out feature dim
#define NP    10016            // rows padded to multiple of 32
#define KB    (NP / 32)        // 313 K-blocks of 32 for the big GEMM
#define MROWS 128              // output rows per workgroup (8 waves x 16)
#define NBLK  ((NN + MROWS - 1) / MROWS)   // 79

// ---- f32 -> bf16 pack (native v_cvt_pk_bf16_f32 when available) ------------
__device__ __forceinline__ uint32_t f2bf1(float f) {
  uint32_t u = __float_as_uint(f);
  return (u + 0x7FFFu + ((u >> 16) & 1u)) >> 16;
}
__device__ __forceinline__ uint32_t f2bf_pk(float lo, float hi) {
#if defined(__has_builtin)
#if __has_builtin(__builtin_amdgcn_cvt_pk_bf16_f32)
  v2bf r = __builtin_amdgcn_cvt_pk_bf16_f32(lo, hi);
  uint32_t u;
  __builtin_memcpy(&u, &r, 4);
  return u;
#else
  return (f2bf1(lo) & 0xFFFFu) | (f2bf1(hi) << 16);
#endif
#else
  return (f2bf1(lo) & 0xFFFFu) | (f2bf1(hi) << 16);
#endif
}

// ============================================================================
// Kernel 1: Y = X @ W_nei^T  (bf16, stored in WMMA B-fragment layout)
//           S = X @ W_self^T (f32, row-major)
// ============================================================================
__global__ __launch_bounds__(256)
void nl_prep_kernel(const float* __restrict__ X,
                    const float* __restrict__ Wn,
                    const float* __restrict__ Ws,
                    uint16_t* __restrict__ yfrag,
                    float* __restrict__ S) {
  __shared__ float Xs[32 * CDIM];        // 16 KB
  __shared__ float Wt[CDIM * 129];       // 66 KB transposed + padded
  const int t  = threadIdx.x;
  const int b  = blockIdx.x;             // K-block index == yfrag kb
  const int i0 = b * 32;

  #pragma unroll
  for (int it = 0; it < 16; ++it) {
    int idx = t + it * 256;
    int r = idx >> 7, k = idx & 127;
    int row = i0 + r;
    Xs[idx] = (row < NN) ? X[(size_t)row * CDIM + k] : 0.0f;
  }
  #pragma unroll
  for (int it = 0; it < 64; ++it) {
    int idx = t + it * 256;
    Wt[(idx & 127) * 129 + (idx >> 7)] = Wn[idx];
  }
  __syncthreads();

  const int o = t & 127;
  const int g = t >> 7;

  float acc[16];
  #pragma unroll
  for (int j = 0; j < 16; ++j) acc[j] = 0.0f;
  for (int k0 = 0; k0 < CDIM; k0 += 8) {
    float w[8];
    #pragma unroll
    for (int kk = 0; kk < 8; ++kk) w[kk] = Wt[(k0 + kk) * 129 + o];
    #pragma unroll
    for (int j = 0; j < 16; ++j)
      #pragma unroll
      for (int kk = 0; kk < 8; ++kk)
        acc[j] = fmaf(Xs[(g * 16 + j) * CDIM + k0 + kk], w[kk], acc[j]);
  }
  // B-fragment layout: lane = 16*K_half + n, element e holds K = e + 16*K_half.
  {
    const int nt = o >> 4;
    const int l  = (g << 4) | (o & 15);
    #pragma unroll
    for (int j = 0; j < 16; ++j)
      yfrag[(size_t)(((b * 8 + nt) * 32 + l) * 16 + j)] = (uint16_t)f2bf1(acc[j]);
  }
  __syncthreads();

  #pragma unroll
  for (int it = 0; it < 64; ++it) {
    int idx = t + it * 256;
    Wt[(idx & 127) * 129 + (idx >> 7)] = Ws[idx];
  }
  __syncthreads();
  #pragma unroll
  for (int j = 0; j < 16; ++j) acc[j] = 0.0f;
  for (int k0 = 0; k0 < CDIM; k0 += 8) {
    float w[8];
    #pragma unroll
    for (int kk = 0; kk < 8; ++kk) w[kk] = Wt[(k0 + kk) * 129 + o];
    #pragma unroll
    for (int j = 0; j < 16; ++j)
      #pragma unroll
      for (int kk = 0; kk < 8; ++kk)
        acc[j] = fmaf(Xs[(g * 16 + j) * CDIM + k0 + kk], w[kk], acc[j]);
  }
  #pragma unroll
  for (int j = 0; j < 16; ++j)
    S[(size_t)(i0 + g * 16 + j) * CDIM + o] = acc[j];
}

// ============================================================================
// Tensor Data Mover: DMA an 8 KB contiguous yfrag block into LDS.
// D# per CDNA5 ISA: count=1, type=2, data_size=8B,
// 1-D tile: tile_dim0 = tensor_dim0 = 1024 units (8192 bytes).
// ============================================================================
__device__ __forceinline__ void tdm_load_b(const uint16_t* src, uint32_t lds_off) {
  uint64_t ga = (uint64_t)(uintptr_t)src;
  v4u g0;
  g0.x = 1u;                                            // count=1, user descriptor
  g0.y = lds_off;                                       // lds_addr (bytes)
  g0.z = (uint32_t)ga;                                  // global_addr[31:0]
  g0.w = ((uint32_t)(ga >> 32) & 0x01FFFFFFu) | (2u << 30);  // addr[56:32] | type=2
  v8i_ g1;
  g1[0] = (3 << 16);           // workgroup_mask=0, data_size=3 (8-byte units)
  g1[1] = (1024 << 16);        // tensor_dim0[15:0]=1024 (bits 79:48)
  g1[2] = (1 << 16);           // tensor_dim0[31:16]=0, tensor_dim1[15:0]=1
  g1[3] = (1024 << 16);        // tensor_dim1[31:16]=0, tile_dim0=1024 (bits 127:112)
  g1[4] = 1;                   // tile_dim1=1, tile_dim2=0
  g1[5] = 1024;                // tensor_dim0_stride[31:0]
  g1[6] = (1024 << 16);        // stride0[47:32]=0, tensor_dim1_stride[15:0]=1024
  g1[7] = 0;
  v4i_ z4 = {0, 0, 0, 0};
#if __clang_major__ >= 23
  v8i_ z8 = {0, 0, 0, 0, 0, 0, 0, 0};
  __builtin_amdgcn_tensor_load_to_lds(g0, g1, z4, z4, z8, 0);
#else
  __builtin_amdgcn_tensor_load_to_lds(g0, g1, z4, z4, 0);
#endif
}

// ---- adj tile staging helpers ----------------------------------------------
__device__ __forceinline__ void load_adj_tile(const float* __restrict__ adj,
                                              int rb0, int c0, int lr, int lc,
                                              float4 st[4]) {
  #pragma unroll
  for (int it = 0; it < 4; ++it) {
    int row = rb0 + lr + it * 32;
    int col = c0 + lc;
    float4 v = {0.0f, 0.0f, 0.0f, 0.0f};
    if (row < NN) {
      const float* p = adj + (size_t)row * NN + col;
      if (col + 3 < NN) {
        v = *(const float4*)p;             // 16B aligned (NN % 4 == 0)
      } else {
        if (col     < NN) v.x = p[0];
        if (col + 1 < NN) v.y = p[1];
        if (col + 2 < NN) v.z = p[2];
      }
    }
    st[it] = v;
  }
}
__device__ __forceinline__ void store_afrag(uint16_t* lds, const float4 st[4],
                                            int lr, int half, int le) {
  #pragma unroll
  for (int it = 0; it < 4; ++it) {
    int r   = lr + it * 32;
    int mt  = r >> 4;
    int l   = (r & 15) | (half << 4);
    int idx = (mt * 32 + l) * 16 + le;     // le % 4 == 0 -> 8B aligned
    *(uint32_t*)&lds[idx]     = f2bf_pk(st[it].x, st[it].y);
    *(uint32_t*)&lds[idx + 2] = f2bf_pk(st[it].z, st[it].w);
  }
}

// ============================================================================
// Kernel 2: out = tanh(adj @ Y + S).  128x128 per WG, double-buffered LDS,
// TDM stages B-fragments, one barrier per K-step, 8 WMMA per wave per step.
// ============================================================================
__global__ __launch_bounds__(256)
void nl_main_kernel(const float* __restrict__ adj,
                    const uint16_t* __restrict__ yfrag,
                    const float* __restrict__ S,
                    float* __restrict__ out) {
  __shared__ __align__(32) uint16_t ldsA[2][8 * 32 * 16];   // 2 x 8 KB A-frags
  __shared__ __align__(32) uint16_t ldsB[2][8 * 32 * 16];   // 2 x 8 KB B-frags
  const int t    = threadIdx.x;
  const int lane = t & 31;               // wave32
  const int wv   = t >> 5;               // 8 waves, wave wv owns M-tile wv
  const int rb0  = blockIdx.x * MROWS;

  v8f acc[8];
  #pragma unroll
  for (int nt = 0; nt < 8; ++nt)
    #pragma unroll
    for (int i = 0; i < 8; ++i) acc[nt][i] = 0.0f;

  const int lr   = t >> 3;                         // tile row 0..31 (+32*it)
  const int lc   = (t & 7) * 4;                    // K within 32-chunk
  const int half = lc >> 4;                        // A-frag lane half
  const int le   = lc & 15;                        // A-frag element

  // ---- prologue: stage tile 0 into buffer 0 ----
  {
    float4 st[4];
    load_adj_tile(adj, rb0, 0, lr, lc, st);
    store_afrag(&ldsA[0][0], st, lr, half, le);
    if (wv == 0) {
      tdm_load_b(yfrag, (uint32_t)(uintptr_t)&ldsB[0][0]);
      __builtin_amdgcn_s_wait_tensorcnt(0);
    }
  }
  __syncthreads();

  for (int kb = 0; kb < KB; ++kb) {
    const int p = kb & 1;
    const bool more = (kb + 1) < KB;

    // Issue next tile's adj loads (latency hidden behind this tile's WMMAs).
    float4 st[4];
    if (more) {
      load_adj_tile(adj, rb0, (kb + 1) * 32, lr, lc, st);
      if (wv == 0)
        tdm_load_b(yfrag + (size_t)(kb + 1) * 4096,
                   (uint32_t)(uintptr_t)&ldsB[p ^ 1][0]);
      // warm L2 two tiles ahead
      if (kb + 2 < KB) {
        int prow = rb0 + (t >> 1);
        if (prow < NN)
          __builtin_prefetch(adj + (size_t)prow * NN + (kb + 2) * 32 + (t & 1) * 16, 0, 1);
      }
    }

    // Compute on buffer p. Load B-frags in groups of 4 so the ds_loads clause
    // together and their latency is covered by the previous group's WMMAs.
    v16bf a = *(const v16bf*)&ldsA[p][(wv * 32 + lane) * 16];
    #pragma unroll
    for (int grp = 0; grp < 2; ++grp) {
      v16bf bq[4];
      #pragma unroll
      for (int i = 0; i < 4; ++i)
        bq[i] = *(const v16bf*)&ldsB[p][((grp * 4 + i) * 32 + lane) * 16];
      #pragma unroll
      for (int i = 0; i < 4; ++i)
        acc[grp * 4 + i] = __builtin_amdgcn_wmma_f32_16x16x32_bf16(
            /*neg_a=*/false, a, /*neg_b=*/false, bq[i],
            /*c_mod=*/(short)0, acc[grp * 4 + i],
            /*reuse_a=*/false, /*reuse_b=*/false);
    }

    // Convert staged adj regs into the other A buffer; wait for TDM; 1 barrier.
    if (more) store_afrag(&ldsA[p ^ 1][0], st, lr, half, le);
    if (wv == 0) __builtin_amdgcn_s_wait_tensorcnt(0);
    __syncthreads();
  }

  // Epilogue: C/D layout -> row = 8*(lane>>4)+vgpr, col = nt*16 + (lane&15).
  const int rbase = rb0 + wv * 16 + ((lane >> 4) << 3);
  const int cb    = lane & 15;
  #pragma unroll
  for (int nt = 0; nt < 8; ++nt) {
    #pragma unroll
    for (int rr = 0; rr < 8; ++rr) {
      int row = rbase + rr;
      if (row < NN) {
        size_t off = (size_t)row * CDIM + (nt * 16 + cb);
        out[off] = tanhf(acc[nt][rr] + S[off]);
      }
    }
  }
}

// ============================================================================
extern "C" void kernel_launch(void* const* d_in, const int* in_sizes, int n_in,
                              void* d_out, int out_size, void* d_ws, size_t ws_size,
                              hipStream_t stream) {
  const float* adj = (const float*)d_in[0];   // [NN, NN]
  const float* X   = (const float*)d_in[1];   // [NN, CDIM]
  const float* Wn  = (const float*)d_in[2];   // [CDIM, CDIM]
  const float* Ws  = (const float*)d_in[3];   // [CDIM, CDIM]
  float* out = (float*)d_out;

  // Workspace: Y fragments (bf16, 2.56 MB) then S (f32, 5.13 MB).
  uint16_t* yfrag = (uint16_t*)d_ws;
  const size_t ybytes = (size_t)KB * 8 * 32 * 16 * sizeof(uint16_t);
  float* S = (float*)((char*)d_ws + ybytes);

  nl_prep_kernel<<<KB, 256, 0, stream>>>(X, Wn, Ws, yfrag, S);
  nl_main_kernel<<<NBLK, 256, 0, stream>>>(adj, yfrag, S, out);
}
